// SVMModel_19267223290147
// MI455X (gfx1250) — matrix operand order repaired and checked by reading, not compiled, not run on hardware
//
#include <hip/hip_runtime.h>

// ---------------------------------------------------------------------------
// RBF-SVM head:  out[b] = bias + sum_n exp(-gamma * ||x_b - t_n||^2) * W[n]
// B=2048, N=32768, D=128.  Cross term via bf16 WMMA (v_wmma_f32_16x16x32_bf16),
// norms kept in f32, exp+reduce fused into the WMMA epilogue.
// ---------------------------------------------------------------------------

#define B_ROWS   2048
#define N_TRAIN  32768
#define DDIM     128
#define NSPLIT   64            // grid.y; each block handles N_TRAIN/NSPLIT = 512 cols
#define NSLICE   (N_TRAIN / NSPLIT)

typedef __attribute__((ext_vector_type(16))) __bf16 v16bf;
typedef __attribute__((ext_vector_type(8)))  float  v8f;

union Frag16 {             // 16 bf16 = 32 bytes
    uint4 u[2];
    v16bf v;
};

// ---- f32 -> bf16 round-to-nearest-even ------------------------------------
__device__ inline unsigned bf16_rne(float f) {
    unsigned u = __float_as_uint(f);
    return (u + 0x7FFFu + ((u >> 16) & 1u)) >> 16;
}
__device__ inline unsigned pack_bf16x2(float lo, float hi) {
    return bf16_rne(lo) | (bf16_rne(hi) << 16);
}

// ---------------------------------------------------------------------------
// Prep: rows of f32 [rows x 128] -> bf16 rows + f32 sum-of-squares per row.
// One wave per row; each lane owns 4 contiguous floats (float4).
// ---------------------------------------------------------------------------
__global__ void prep_bf16_sq(const float* __restrict__ src,
                             unsigned short* __restrict__ dst,
                             float* __restrict__ sq, int rows) {
    int w    = threadIdx.x >> 5;
    int lane = threadIdx.x & 31;
    int row  = blockIdx.x * 4 + w;
    if (row >= rows) return;

    const float4* p = (const float4*)(src + (size_t)row * DDIM);
    float4 v = p[lane];
    float ss = v.x * v.x + v.y * v.y + v.z * v.z + v.w * v.w;

    uint2 o;
    o.x = pack_bf16x2(v.x, v.y);
    o.y = pack_bf16x2(v.z, v.w);
    ((uint2*)(dst + (size_t)row * DDIM))[lane] = o;

    #pragma unroll
    for (int mask = 16; mask >= 1; mask >>= 1)
        ss += __shfl_xor(ss, mask, 32);
    if (lane == 0) sq[row] = ss;
}

// ---------------------------------------------------------------------------
// Main: one block = 16-row x-tile  x  512-col n-slice.  4 waves, each wave
// does a 16x16 tile per iteration (8 iterations).  A-fragments for the full
// D=128 are register-resident; only B reloads per tile.
// ---------------------------------------------------------------------------
__global__ void __launch_bounds__(128)
rbf_wmma_kernel(const unsigned short* __restrict__ xbf,
                const unsigned short* __restrict__ tbf,
                const float* __restrict__ x2,
                const float* __restrict__ t2,
                const float* __restrict__ W,
                const float* __restrict__ gamma,
                float* __restrict__ partial) {
    const int w       = threadIdx.x >> 5;
    const int lane    = threadIdx.x & 31;
    const int half    = lane >> 4;        // 0 or 1
    const int l16     = lane & 15;
    const int rowbase = blockIdx.x * 16;
    const int nbase0  = blockIdx.y * NSLICE;

    const float g = gamma[0];

    // ---- A fragments: row m = rowbase + l16, all 4 K-steps (D=128) --------
    // 16-bit A 16x32 layout: lane holds elements [32s + 8h .. +7] and [+16..+23]
    Frag16 afr[4];
    {
        const unsigned short* arow = xbf + (size_t)(rowbase + l16) * DDIM;
        #pragma unroll
        for (int s = 0; s < 4; ++s) {
            const uint4* p = (const uint4*)(arow + s * 32 + half * 8);
            afr[s].u[0] = p[0];   // k .. k+7
            afr[s].u[1] = p[2];   // k+16 .. k+23
        }
    }

    // x2 for the 8 rows this lane's C-fragment covers: row = rowbase + i + 8*half
    float x2v[8];
    #pragma unroll
    for (int i = 0; i < 8; ++i)
        x2v[i] = x2[rowbase + half * 8 + i];

    float acc[8];
    #pragma unroll
    for (int i = 0; i < 8; ++i) acc[i] = 0.0f;

    // ---- loop over n-tiles: wave w covers cols nbase0 + it*64 + w*16 ------
    for (int it = 0; it < NSLICE / 64; ++it) {
        const int ncol = nbase0 + it * 64 + w * 16 + l16;
        const unsigned short* brow = tbf + (size_t)ncol * DDIM;

        v8f c = {};
        #pragma unroll
        for (int s = 0; s < 4; ++s) {
            // 16-bit B 32x16 layout: lane holds K [32s+8h .. +7] and [+16..+23]
            Frag16 bfr;
            const uint4* q = (const uint4*)(brow + s * 32 + half * 8);
            bfr.u[0] = q[0];
            bfr.u[1] = q[2];
            c = __builtin_amdgcn_wmma_f32_16x16x32_bf16(
                    false, afr[s].v, false, bfr.v, (short)0, c, false, false);
        }

        const float t2n = t2[ncol];
        const float wn  = W[ncol];
        #pragma unroll
        for (int i = 0; i < 8; ++i) {
            float d2 = x2v[i] + t2n - 2.0f * c[i];
            acc[i] += __expf(-g * d2) * wn;
        }
    }

    // ---- reduce across the 16 lanes of each half (cols of the tile) -------
    #pragma unroll
    for (int i = 0; i < 8; ++i) {
        #pragma unroll
        for (int mask = 8; mask >= 1; mask >>= 1)
            acc[i] += __shfl_xor(acc[i], mask, 32);
    }

    // ---- combine the 4 waves deterministically via LDS --------------------
    __shared__ float red[4][16];
    if (l16 == 0) {
        #pragma unroll
        for (int i = 0; i < 8; ++i)
            red[w][half * 8 + i] = acc[i];
    }
    __syncthreads();

    if (threadIdx.x < 16) {
        float s = red[0][threadIdx.x] + red[1][threadIdx.x] +
                  red[2][threadIdx.x] + red[3][threadIdx.x];
        partial[(size_t)(rowbase + threadIdx.x) * NSPLIT + blockIdx.y] = s;
    }
}

// ---------------------------------------------------------------------------
// Finalize: out[b] = bias + sum_j partial[b][j]   (fixed order -> deterministic)
// ---------------------------------------------------------------------------
__global__ void finalize_kernel(const float* __restrict__ partial,
                                const float* __restrict__ bias,
                                float* __restrict__ out) {
    int b = blockIdx.x * blockDim.x + threadIdx.x;
    if (b < B_ROWS) {
        float s = bias[0];
        const float* p = partial + (size_t)b * NSPLIT;
        #pragma unroll 8
        for (int j = 0; j < NSPLIT; ++j) s += p[j];
        out[b] = s;
    }
}

// ---------------------------------------------------------------------------
extern "C" void kernel_launch(void* const* d_in, const int* in_sizes, int n_in,
                              void* d_out, int out_size, void* d_ws, size_t ws_size,
                              hipStream_t stream) {
    const float* x     = (const float*)d_in[0];   // [B, D]
    const float* train = (const float*)d_in[1];   // [N, D]
    const float* gamma = (const float*)d_in[2];   // [1]
    const float* W     = (const float*)d_in[3];   // [1, N]
    const float* bias  = (const float*)d_in[4];   // [1]
    float* out = (float*)d_out;                   // [B]

    // workspace carve-up (all offsets 256B aligned)
    char* ws = (char*)d_ws;
    unsigned short* xbf = (unsigned short*)ws;                       // 512 KB
    unsigned short* tbf = (unsigned short*)(ws + (size_t)B_ROWS * DDIM * 2);          // 8 MB
    char* after_tbf = ws + (size_t)B_ROWS * DDIM * 2 + (size_t)N_TRAIN * DDIM * 2;
    float* x2      = (float*)after_tbf;                              // 8 KB
    float* t2      = (float*)(after_tbf + (size_t)B_ROWS * 4);       // 128 KB
    float* partial = (float*)(after_tbf + (size_t)B_ROWS * 4 + (size_t)N_TRAIN * 4);  // 512 KB

    // 1) bf16 conversion + row norms
    prep_bf16_sq<<<B_ROWS / 4, 128, 0, stream>>>(x, xbf, x2, B_ROWS);
    prep_bf16_sq<<<N_TRAIN / 4, 128, 0, stream>>>(train, tbf, t2, N_TRAIN);

    // 2) fused WMMA + exp + weighted partial reduction
    dim3 grid(B_ROWS / 16, NSPLIT);
    rbf_wmma_kernel<<<grid, 128, 0, stream>>>(xbf, tbf, x2, t2, W, gamma, partial);

    // 3) deterministic final sum + bias
    finalize_kernel<<<(B_ROWS + 127) / 128, 128, 0, stream>>>(partial, bias, out);
}
